// GSAT_39470749450421
// MI455X (gfx1250) — matrix operand order repaired and checked by reading, not compiled, not run on hardware
//
#include <hip/hip_runtime.h>
#include <hip/hip_bf16.h>
#include <math.h>

#define N_NODES   50000
#define DIM       128
#define E_EDGES   400000
#define HALF_E    200000
#define ROW_TILES (N_NODES / 16)   // 3125
#define EDGE_TILES (E_EDGES / 16)  // 25000
#define ND        ((size_t)N_NODES * DIM)

typedef __attribute__((ext_vector_type(16))) _Float16 v16h;
typedef __attribute__((ext_vector_type(8)))  float    v8f;

__device__ __forceinline__ v8f wmma_f32_16x16x32_f16(v16h a, v16h b, v8f c) {
  // (neg_a, A, neg_b, B, c_mod, C, reuse_a, reuse_b)
  return __builtin_amdgcn_wmma_f32_16x16x32_f16(false, a, false, b, (short)0, c,
                                                false, false);
}

// Load 4 A-fragments (K = 0..127) from an f32 row, vectorized float4 loads.
// ISA 16-bit A 16x32 layout: lanes 0-15 -> M=lane, K {0..7,16..23};
// lanes 16-31 -> M=lane-16, K {8..15,24..31}. koff = (lane>=16)?8:0.
__device__ __forceinline__ void load_afrag_row(const float* __restrict__ rp,
                                               int koff, v16h frag[4]) {
#pragma unroll
  for (int kc = 0; kc < 4; ++kc) {
    const int kb = kc * 32 + koff;
    const float4 a0 = *(const float4*)(rp + kb);
    const float4 a1 = *(const float4*)(rp + kb + 4);
    const float4 a2 = *(const float4*)(rp + kb + 16);
    const float4 a3 = *(const float4*)(rp + kb + 20);
    frag[kc][0]  = (_Float16)a0.x; frag[kc][1]  = (_Float16)a0.y;
    frag[kc][2]  = (_Float16)a0.z; frag[kc][3]  = (_Float16)a0.w;
    frag[kc][4]  = (_Float16)a1.x; frag[kc][5]  = (_Float16)a1.y;
    frag[kc][6]  = (_Float16)a1.z; frag[kc][7]  = (_Float16)a1.w;
    frag[kc][8]  = (_Float16)a2.x; frag[kc][9]  = (_Float16)a2.y;
    frag[kc][10] = (_Float16)a2.z; frag[kc][11] = (_Float16)a2.w;
    frag[kc][12] = (_Float16)a3.x; frag[kc][13] = (_Float16)a3.y;
    frag[kc][14] = (_Float16)a3.z; frag[kc][15] = (_Float16)a3.w;
  }
}

// Stage a 128(K) x 128(N) slab of a row-major f32 weight into LDS,
// pre-swizzled into B-fragment order: frag f = kc*8+nt at
// Bsh[f*512 + lane*16 + j]. B 32x16 f16 layout: lanes 0-15 -> N=lane,
// K=kc*32+{0..7,16..23}; lanes 16-31 -> N=lane-16, K=kc*32+{8..15,24..31}.
__device__ __forceinline__ void stage_B(const float* __restrict__ W, int krow0,
                                        _Float16* __restrict__ Bsh, int tid) {
  for (int idx = tid; idx < 16384; idx += 256) {
    const int j  = idx & 15;
    const int ln = (idx >> 4) & 31;
    const int f  = idx >> 9;
    const int nt = f & 7;
    const int kc = f >> 3;
    const int n  = nt * 16 + (ln & 15);
    const int k  = krow0 + kc * 32 + ((ln >> 4) << 3) + (j & 7) + ((j >> 3) << 4);
    Bsh[idx] = (_Float16)W[k * DIM + n];
  }
}

// ---------------------------------------------------------------------------
// Kernel 1: four node GEMMs sharing one A-tile.
// ---------------------------------------------------------------------------
__global__ __launch_bounds__(256) void node_gemm4(
    const float* __restrict__ x,
    const float* __restrict__ W1, const float* __restrict__ W2,
    const float* __restrict__ Wc1, const float* __restrict__ Wc2,
    float* __restrict__ out_h, float* __restrict__ out_emb,
    float* __restrict__ out_hc, float* __restrict__ out_agc) {
  __shared__ _Float16 Bsh[16384];  // 32 KB
  const int tid  = threadIdx.x;
  const int wave = tid >> 5;
  const int lane = tid & 31;
  const int tile = blockIdx.x * 8 + wave;
  const bool active = (tile < ROW_TILES);   // wave-uniform -> EXEC stays full
  const int m     = lane & 15;
  const int koff  = (lane >> 4) << 3;
  const int rbase = (lane >> 4) << 3;       // C: VGPR r -> M = r + rbase

  v16h afrag[4];
  if (active) {
    const float* rp = x + (size_t)(tile * 16 + m) * DIM;
    load_afrag_row(rp, koff, afrag);
  }

  const float* Ws[4]   = {W1, W2, Wc1, Wc2};
  float*       Outs[4] = {out_h, out_emb, out_hc, out_agc};

  for (int w = 0; w < 4; ++w) {
    __syncthreads();                 // WAR on Bsh vs previous weight
    stage_B(Ws[w], 0, Bsh, tid);
    __syncthreads();
    if (active) {
      float* O = Outs[w];
#pragma unroll
      for (int nt = 0; nt < 8; ++nt) {
        v8f acc = {};
#pragma unroll
        for (int kc = 0; kc < 4; ++kc) {
          v16h b = *(const v16h*)(&Bsh[(kc * 8 + nt) * 512 + lane * 16]);
          acc = wmma_f32_16x16x32_f16(afrag[kc], b, acc);
        }
        const int col = nt * 16 + (lane & 15);
#pragma unroll
        for (int r = 0; r < 8; ++r)
          O[(size_t)(tile * 16 + rbase + r) * DIM + col] = acc[r];
      }
    }
  }
}

// ---------------------------------------------------------------------------
// CSR construction: histogram -> scan -> bucket (atomic-free aggregation after)
// ---------------------------------------------------------------------------
__global__ __launch_bounds__(256) void zero_ints(int* __restrict__ p, int n) {
  const int i = blockIdx.x * 256 + threadIdx.x;
  if (i < n) p[i] = 0;
}

__global__ __launch_bounds__(256) void hist_dst(const int* __restrict__ dst,
                                                int* __restrict__ counts) {
  const int e = blockIdx.x * 256 + threadIdx.x;
  if (e < E_EDGES) atomicAdd(&counts[dst[e]], 1);
}

// Single-block exclusive scan of counts[N] -> row_start[N+1] and cursor[N].
__global__ __launch_bounds__(1024) void scan_counts(
    const int* __restrict__ counts, int* __restrict__ row_start,
    int* __restrict__ cursor) {
  __shared__ int sh[1024];
  __shared__ int carry;
  const int tid = threadIdx.x;
  if (tid == 0) carry = 0;
  __syncthreads();
  for (int base = 0; base < N_NODES; base += 1024) {
    const int i = base + tid;
    const int v = (i < N_NODES) ? counts[i] : 0;
    sh[tid] = v;
    __syncthreads();
#pragma unroll
    for (int off = 1; off < 1024; off <<= 1) {
      const int t = (tid >= off) ? sh[tid - off] : 0;
      __syncthreads();
      sh[tid] += t;
      __syncthreads();
    }
    const int excl = sh[tid] - v;
    if (i < N_NODES) {
      const int rs = carry + excl;
      row_start[i] = rs;
      cursor[i] = rs;
    }
    __syncthreads();                   // all reads of carry done
    if (tid == 1023) carry += sh[1023];
    __syncthreads();                   // carry visible; sh reusable
  }
  if (tid == 0) row_start[N_NODES] = carry;
}

__global__ __launch_bounds__(256) void bucket_edges(
    const int* __restrict__ dst, int* __restrict__ cursor,
    int* __restrict__ eids) {
  const int e = blockIdx.x * 256 + threadIdx.x;
  if (e < E_EDGES) {
    const int pos = atomicAdd(&cursor[dst[e]], 1);
    eids[pos] = e;
  }
}

// ---------------------------------------------------------------------------
// Kernel: emb[n] = relu(base[n] + sum_{e: dst[e]==n} h[src[e]])
// One wave per node, atomic-free, register accumulation (float4 per lane).
// ---------------------------------------------------------------------------
__global__ __launch_bounds__(256) void aggregate_emb(
    const float* __restrict__ h, const int* __restrict__ row_start,
    const int* __restrict__ eids, const int* __restrict__ src,
    float* __restrict__ emb) {
  const int n = blockIdx.x * 8 + (threadIdx.x >> 5);
  if (n >= N_NODES) return;
  const int lane = threadIdx.x & 31;
  float4 acc = ((const float4*)(emb + (size_t)n * DIM))[lane];  // x@W2
  const int b0 = row_start[n], b1 = row_start[n + 1];
  for (int i = b0; i < b1; ++i) {
    const int e = eids[i];
    if (i + 1 < b1)
      __builtin_prefetch(h + (size_t)src[eids[i + 1]] * DIM + lane * 4, 0, 1);
    const float4 v = ((const float4*)(h + (size_t)src[e] * DIM))[lane];
    acc.x += v.x; acc.y += v.y; acc.z += v.z; acc.w += v.w;
  }
  acc.x = fmaxf(acc.x, 0.f); acc.y = fmaxf(acc.y, 0.f);
  acc.z = fmaxf(acc.z, 0.f); acc.w = fmaxf(acc.w, 0.f);
  ((float4*)(emb + (size_t)n * DIM))[lane] = acc;
}

// ---------------------------------------------------------------------------
// Kernel: att[e] = sigmoid(tanh(concat(emb[src],emb[dst]) @ We1) . we2
//                          + gumbel(noise_u[e]))   — 16 edges per wave.
// ---------------------------------------------------------------------------
__global__ __launch_bounds__(256) void edge_att_kernel(
    const float* __restrict__ emb, const float* __restrict__ We1,
    const float* __restrict__ we2, const float* __restrict__ noise_u,
    const int* __restrict__ src, const int* __restrict__ dst,
    float* __restrict__ att) {
  __shared__ _Float16 Bsh[16384];   // 32 KB, one K-half of We1
  __shared__ float logit_sh[8][16];
  const int tid   = threadIdx.x;
  const int wave  = tid >> 5;
  const int lane  = tid & 31;
  const int etile = blockIdx.x * 8 + wave;   // exact: 3125 blocks * 8
  const int m     = lane & 15;
  const int koff  = (lane >> 4) << 3;
  const int e     = etile * 16 + m;

  if (lane < 16) logit_sh[wave][lane] = 0.f;

  v8f acc[8];
#pragma unroll
  for (int nt = 0; nt < 8; ++nt) acc[nt] = (v8f){};

  for (int khalf = 0; khalf < 2; ++khalf) {
    __syncthreads();
    stage_B(We1, khalf * DIM, Bsh, tid);
    __syncthreads();
    const int row = (khalf == 0) ? src[e] : dst[e];
    const float* rp = emb + (size_t)row * DIM;
    v16h afrag[4];
    load_afrag_row(rp, koff, afrag);
#pragma unroll
    for (int nt = 0; nt < 8; ++nt) {
#pragma unroll
      for (int kc = 0; kc < 4; ++kc) {
        v16h b = *(const v16h*)(&Bsh[(kc * 8 + nt) * 512 + lane * 16]);
        acc[nt] = wmma_f32_16x16x32_f16(afrag[kc], b, acc[nt]);
      }
    }
  }

  // tanh, multiply by we2[n], reduce over n (16 lanes per output row).
  float partial[8];
#pragma unroll
  for (int r = 0; r < 8; ++r) partial[r] = 0.f;
#pragma unroll
  for (int nt = 0; nt < 8; ++nt) {
    const float wn = we2[nt * 16 + (lane & 15)];
#pragma unroll
    for (int r = 0; r < 8; ++r) partial[r] += tanhf(acc[nt][r]) * wn;
  }
  const int rbase = (lane >> 4) << 3;
#pragma unroll
  for (int r = 0; r < 8; ++r)
    atomicAdd(&logit_sh[wave][rbase + r], partial[r]);
  __syncthreads();

  if (lane < 16) {
    const float lg = logit_sh[wave][lane];
    const float u  = noise_u[e];
    const float ns = logf(u) - log1pf(-u);
    att[e] = 1.f / (1.f + expf(-(lg + ns)));
  }
}

// ---------------------------------------------------------------------------
// Kernel: out[n] = relu(out[n] + sum_{e: dst[e]==n} 0.5*(att[e]+att[rev]) *
//                       hc[src[e]])   — one wave per node, atomic-free.
// ---------------------------------------------------------------------------
__global__ __launch_bounds__(256) void aggregate_out(
    const float* __restrict__ hc, const float* __restrict__ att,
    const int* __restrict__ row_start, const int* __restrict__ eids,
    const int* __restrict__ src, float* __restrict__ out) {
  const int n = blockIdx.x * 8 + (threadIdx.x >> 5);
  if (n >= N_NODES) return;
  const int lane = threadIdx.x & 31;
  float4 acc = ((const float4*)(out + (size_t)n * DIM))[lane];  // x@Wc2
  const int b0 = row_start[n], b1 = row_start[n + 1];
  for (int i = b0; i < b1; ++i) {
    const int e = eids[i];
    if (i + 1 < b1)
      __builtin_prefetch(hc + (size_t)src[eids[i + 1]] * DIM + lane * 4, 0, 1);
    const int partner = (e < HALF_E) ? (e + HALF_E) : (e - HALF_E);
    const float ea = 0.5f * (att[e] + att[partner]);
    const float4 v = ((const float4*)(hc + (size_t)src[e] * DIM))[lane];
    acc.x += ea * v.x; acc.y += ea * v.y;
    acc.z += ea * v.z; acc.w += ea * v.w;
  }
  acc.x = fmaxf(acc.x, 0.f); acc.y = fmaxf(acc.y, 0.f);
  acc.z = fmaxf(acc.z, 0.f); acc.w = fmaxf(acc.w, 0.f);
  ((float4*)(out + (size_t)n * DIM))[lane] = acc;
}

extern "C" void kernel_launch(void* const* d_in, const int* in_sizes, int n_in,
                              void* d_out, int out_size, void* d_ws,
                              size_t ws_size, hipStream_t stream) {
  (void)in_sizes; (void)n_in; (void)out_size; (void)ws_size;
  const float* x       = (const float*)d_in[0];
  const float* W1      = (const float*)d_in[1];
  const float* W2      = (const float*)d_in[2];
  const float* We1     = (const float*)d_in[3];
  const float* we2     = (const float*)d_in[4];
  const float* Wc1     = (const float*)d_in[5];
  const float* Wc2     = (const float*)d_in[6];
  const float* noise_u = (const float*)d_in[7];
  const int*   ei      = (const int*)d_in[8];
  const int*   src     = ei;
  const int*   dst     = ei + E_EDGES;
  float* out = (float*)d_out;

  float* ws     = (float*)d_ws;
  float* ws_h   = ws;            // [N, D]
  float* ws_emb = ws + 1 * ND;   // [N, D]
  float* ws_hc  = ws + 2 * ND;   // [N, D]
  float* ws_att = ws + 3 * ND;   // [E]
  int* wsi       = (int*)(ws + 3 * ND + E_EDGES);
  int* counts    = wsi;                      // [N]
  int* cursor    = wsi + N_NODES;            // [N]
  int* row_start = wsi + 2 * N_NODES;        // [N+1]
  int* eids      = wsi + 3 * N_NODES + 1;    // [E]

  // 1) h = x@W1 ; emb0 = x@W2 ; hc = x@Wc1 ; d_out = x@Wc2
  node_gemm4<<<(ROW_TILES + 7) / 8, 256, 0, stream>>>(
      x, W1, W2, Wc1, Wc2, ws_h, ws_emb, ws_hc, out);
  // 2) CSR by dst (built each call; reused by both aggregation passes)
  zero_ints<<<(N_NODES + 255) / 256, 256, 0, stream>>>(counts, N_NODES);
  hist_dst<<<(E_EDGES + 255) / 256, 256, 0, stream>>>(dst, counts);
  scan_counts<<<1, 1024, 0, stream>>>(counts, row_start, cursor);
  bucket_edges<<<(E_EDGES + 255) / 256, 256, 0, stream>>>(dst, cursor, eids);
  // 3) emb = relu(x@W2 + gather-sum h[src])  (atomic-free, fused relu)
  aggregate_emb<<<(N_NODES + 7) / 8, 256, 0, stream>>>(
      ws_h, row_start, eids, src, ws_emb);
  // 4) att = sigmoid(tanh(concat(emb[src],emb[dst])@We1).we2 + gumbel)
  edge_att_kernel<<<EDGE_TILES / 8, 256, 0, stream>>>(
      ws_emb, We1, we2, noise_u, src, dst, ws_att);
  // 5) d_out = relu(x@Wc2 + gather-sum edge_att*hc[src])
  aggregate_out<<<(N_NODES + 7) / 8, 256, 0, stream>>>(
      ws_hc, ws_att, row_start, eids, src, out);
}
